// SingleLayerGRU_65841848648479
// MI455X (gfx1250) — compile-verified, matrix operand order checked by reading, not run on hardware
//
#include <hip/hip_runtime.h>

// ---------------------------------------------------------------------------
// Single-layer GRU (D=256, H=512, O=256, B=256, T=512) for gfx1250 (MI455X).
//
// Strategy (see analysis): recurrence is independent per batch row ->
// one workgroup (512 threads = 16 wave32) owns a 16-row batch tile and keeps
// h (16x512 bf16) resident in LDS for all 512 timesteps. No inter-WG sync.
// gi is computed on the fly per step (no 805MB gi_all round-trip).
// GEMMs use v_wmma_f32_16x16x32_bf16 (bf16 operands, f32 accumulate).
// Weights are pre-packed once into B-fragment order in workspace (~2.5 MB).
// ---------------------------------------------------------------------------

typedef __attribute__((ext_vector_type(16))) __bf16 v16bf;
typedef __attribute__((ext_vector_type(8)))  float  v8f;

namespace gru {
constexpr int D = 256, H = 512, O = 256, Bsz = 256, T = 512;
constexpr int G = 3 * H;                 // 1536 gate columns
constexpr int BT = 16;                   // batch tile rows (WMMA M)
constexpr int WAVES = 16, THREADS = WAVES * 32;
constexpr int XP = D + 10;               // padded LDS row strides (bank-conflict free)
constexpr int HP = H + 10;
constexpr int KT_IH = D / 32;            // 8
constexpr int KT_HH = H / 32;            // 16
constexpr int NTRIP = 32;                // (r,z,n) column triples = 512/16
constexpr int TPW = NTRIP / WAVES;       // 2 triples per wave
}

__device__ __forceinline__ unsigned short f2bf(float f) {
  unsigned u = __builtin_bit_cast(unsigned, f);
  unsigned r = (u + 0x7FFFu + ((u >> 16) & 1u)) >> 16;   // round-to-nearest-even
  return (unsigned short)r;
}
__device__ __forceinline__ float bf2f(unsigned short s) {
  unsigned u = ((unsigned)s) << 16;
  return __builtin_bit_cast(float, u);
}

union AFrag { v16bf v; unsigned u[8]; };
union BFrag { v16bf v; uint4 q[2]; };

// A fragment (16x32 bf16, ISA layout): lanes 0-15 rows M=0..15;
// VGPR0-3 hold K = (0..7)+8*hi ; VGPR4-7 hold K = (16..23)+8*hi  (hi = lane/16)
__device__ __forceinline__ void load_a(AFrag& a, const unsigned short* row,
                                       int kbase, int hi) {
#pragma unroll
  for (int jj = 0; jj < 8; ++jj) {
    int k = kbase + 8 * hi + ((jj < 4) ? 2 * jj : 16 + 2 * (jj - 4));
    a.u[jj] = *(const unsigned*)(row + k);      // 4B-aligned LDS dword load
  }
}

// B fragment: packed at weight-conversion time, one v16bf (32B) per lane.
__device__ __forceinline__ v16bf load_b(const unsigned short* __restrict__ pack,
                                        int nt, int kt, int KT, int lane) {
  const uint4* p = (const uint4*)(pack + (((size_t)nt * KT + kt) * 32 + lane) * 16);
  BFrag b; b.q[0] = p[0]; b.q[1] = p[1];
  return b.v;
}

__device__ __forceinline__ v8f wmma_bf16(v8f c, v16bf a, v16bf b) {
  return __builtin_amdgcn_wmma_f32_16x16x32_bf16(false, a, false, b,
                                                 (short)0, c, false, false);
}

__device__ __forceinline__ float sigmoidf_(float x) {
  return 1.0f / (1.0f + __expf(-x));
}

// ---------------------------------------------------------------------------
// Pack W (row-major [N][K] f32, used as B = W^T in GEMM) into bf16 fragments:
// dst[((nt*KT + kt)*32 + lane)*16 + e], lane -> column nt*16 + lane%16,
// element e -> K = kt*32 + 8*(lane/16) + (e<8 ? e : 16+(e-8)).
// Mirrors the A-matrix K striping (self-consistent with load_b).
// ---------------------------------------------------------------------------
__global__ void gru_pack_b(const float* __restrict__ W,
                           unsigned short* __restrict__ dst, int N, int K) {
  int p = blockIdx.x * blockDim.x + threadIdx.x;
  int KT = K / 32;
  int total = (N / 16) * KT * 32;
  if (p >= total) return;
  int lane = p & 31;
  int kt = (p >> 5) % KT;
  int nt = p / (32 * KT);
  int n = nt * 16 + (lane & 15);
  int hi = lane >> 4;
  unsigned short* o = dst + (size_t)p * 16;
#pragma unroll
  for (int e = 0; e < 16; ++e) {
    int k = kt * 32 + 8 * hi + ((e < 8) ? e : 16 + (e - 8));
    o[e] = f2bf(W[(size_t)n * K + k]);
  }
}

// ---------------------------------------------------------------------------
// Main persistent GRU kernel: grid = B/16 blocks, 512 threads (16 waves).
// ---------------------------------------------------------------------------
__global__ __launch_bounds__(gru::THREADS)
void gru_main(const float* __restrict__ x,
              const float* __restrict__ b_ih, const float* __restrict__ b_hh,
              const float* __restrict__ b_out,
              const unsigned short* __restrict__ wih_p,
              const unsigned short* __restrict__ whh_p,
              const unsigned short* __restrict__ wout_p,
              float* __restrict__ out) {
  using namespace gru;
  __shared__ unsigned short hs[BT * HP];   // resident h tile, bf16
  __shared__ unsigned short xs[BT * XP];   // staged x[:, t, :] tile, bf16

  const int tid = threadIdx.x;
  const int wave = tid >> 5, lane = tid & 31;
  const int m = lane & 15;                 // row within tile (A layout), column (C layout)
  const int hi = lane >> 4;
  const int b0 = blockIdx.x * BT;
  const int trip0 = wave * TPW;            // this wave's gate-column triples

  for (int i = tid; i < BT * HP; i += THREADS) hs[i] = 0;   // h0 = 0

  // Per-lane bias scalars (C layout: all 8 acc elements share column).
  float bR[TPW], bZ[TPW], bNi[TPW], bNh[TPW];
#pragma unroll
  for (int j = 0; j < TPW; ++j) {
    int col = (trip0 + j) * 16 + m;
    bR[j]  = b_ih[col] + b_hh[col];
    bZ[j]  = b_ih[col + H] + b_hh[col + H];
    bNi[j] = b_ih[col + 2 * H];            // n gate: keep gi / gh separate
    bNh[j] = b_hh[col + 2 * H];
  }
  __syncthreads();

  for (int t = 0; t < T; ++t) {
    // Stage x[b0:b0+16, t, :] -> LDS bf16 (cooperative, 8 elems/thread).
    for (int i = tid; i < BT * D; i += THREADS) {
      int r = i >> 8, d = i & (D - 1);
      xs[r * XP + d] = f2bf(x[(size_t)(b0 + r) * T * D + (size_t)t * D + d]);
    }
    __syncthreads();

    v8f accR[TPW], accZ[TPW], accNi[TPW], accNh[TPW];
#pragma unroll
    for (int j = 0; j < TPW; ++j) { accR[j] = {}; accZ[j] = {}; accNi[j] = {}; accNh[j] = {}; }

    // gi contribution: K over D (A from xs, B from packed W_ih).
#pragma unroll 2
    for (int kt = 0; kt < KT_IH; ++kt) {
      AFrag a; load_a(a, &xs[m * XP], kt * 32, hi);
#pragma unroll
      for (int j = 0; j < TPW; ++j) {
        int c = trip0 + j;
        accR[j]  = wmma_bf16(accR[j],  a.v, load_b(wih_p, c,      kt, KT_IH, lane));
        accZ[j]  = wmma_bf16(accZ[j],  a.v, load_b(wih_p, c + 32, kt, KT_IH, lane));
        accNi[j] = wmma_bf16(accNi[j], a.v, load_b(wih_p, c + 64, kt, KT_IH, lane));
      }
    }
    // gh contribution: K over H (A from resident hs, B from packed W_hh).
#pragma unroll 2
    for (int kt = 0; kt < KT_HH; ++kt) {
      AFrag a; load_a(a, &hs[m * HP], kt * 32, hi);
#pragma unroll
      for (int j = 0; j < TPW; ++j) {
        int c = trip0 + j;
        accR[j]  = wmma_bf16(accR[j],  a.v, load_b(whh_p, c,      kt, KT_HH, lane));
        accZ[j]  = wmma_bf16(accZ[j],  a.v, load_b(whh_p, c + 32, kt, KT_HH, lane));
        accNh[j] = wmma_bf16(accNh[j], a.v, load_b(whh_p, c + 64, kt, KT_HH, lane));
      }
    }

    // Gate math + h update, entirely in registers (f32).
    unsigned short hnew[TPW][8];
#pragma unroll
    for (int j = 0; j < TPW; ++j) {
      int hcol = (trip0 + j) * 16 + m;
#pragma unroll
      for (int e = 0; e < 8; ++e) {
        int row = e + 8 * hi;
        float r  = sigmoidf_(accR[j][e] + bR[j]);
        float z  = sigmoidf_(accZ[j][e] + bZ[j]);
        float nn = tanhf(accNi[j][e] + bNi[j] + r * (accNh[j][e] + bNh[j]));
        float hp = bf2f(hs[row * HP + hcol]);
        hnew[j][e] = f2bf((1.0f - z) * nn + z * hp);
      }
    }
    __syncthreads();   // all waves done reading hs for this step
#pragma unroll
    for (int j = 0; j < TPW; ++j) {
      int hcol = (trip0 + j) * 16 + m;
#pragma unroll
      for (int e = 0; e < 8; ++e) hs[(e + 8 * hi) * HP + hcol] = hnew[j][e];
    }
    __syncthreads();   // h_new visible to all waves for step t+1
  }

  // Final projection: out = h_T @ W_out^T + b_out  (O=256 -> 1 N-tile / wave)
  {
    int nt = wave;
    v8f acc = {};
#pragma unroll 2
    for (int kt = 0; kt < KT_HH; ++kt) {
      AFrag a; load_a(a, &hs[m * HP], kt * 32, hi);
      acc = wmma_bf16(acc, a.v, load_b(wout_p, nt, kt, KT_HH, lane));
    }
    float bo = b_out[nt * 16 + m];
#pragma unroll
    for (int e = 0; e < 8; ++e) {
      int row = e + 8 * hi;
      out[(size_t)(b0 + row) * O + nt * 16 + m] = acc[e] + bo;
    }
  }
}

extern "C" void kernel_launch(void* const* d_in, const int* in_sizes, int n_in,
                              void* d_out, int out_size, void* d_ws, size_t ws_size,
                              hipStream_t stream) {
  using namespace gru;
  const float* x     = (const float*)d_in[0];   // [B,T,D]
  const float* W_ih  = (const float*)d_in[1];   // [3H,D]
  const float* W_hh  = (const float*)d_in[2];   // [3H,H]
  const float* b_ih  = (const float*)d_in[3];   // [3H]
  const float* b_hh  = (const float*)d_in[4];   // [3H]
  const float* W_out = (const float*)d_in[5];   // [O,H]
  const float* b_out = (const float*)d_in[6];   // [O]
  float* out = (float*)d_out;                   // [B,O]

  // Workspace: packed bf16 weight fragments (deterministic, rebuilt each call).
  // Needs (3H*D + 3H*H + O*H)*2 = 2.5 MB.
  unsigned short* wih_p  = (unsigned short*)d_ws;            // 3H x D
  unsigned short* whh_p  = wih_p + (size_t)G * D;            // 3H x H
  unsigned short* wout_p = whh_p + (size_t)G * H;            // O  x H

  {
    int tot = (G / 16) * (D / 32) * 32;
    gru_pack_b<<<(tot + 255) / 256, 256, 0, stream>>>(W_ih, wih_p, G, D);
  }
  {
    int tot = (G / 16) * (H / 32) * 32;
    gru_pack_b<<<(tot + 255) / 256, 256, 0, stream>>>(W_hh, whh_p, G, H);
  }
  {
    int tot = (O / 16) * (H / 32) * 32;
    gru_pack_b<<<(tot + 255) / 256, 256, 0, stream>>>(W_out, wout_p, O, H);
  }
  gru_main<<<Bsz / BT, THREADS, 0, stream>>>(x, b_ih, b_hh, b_out,
                                             wih_p, whh_p, wout_p, out);
}
